// MultiHeadAttention_69741678952755
// MI455X (gfx1250) — compile-verified, matrix operand order checked by reading, not run on hardware
//
#include <hip/hip_runtime.h>
#include <hip/hip_bf16.h>

#define B_  2
#define S_  4096
#define D_  768
#define H_  12
#define DH_ 64

typedef _Float16 v16h __attribute__((ext_vector_type(16)));
typedef float    v8f  __attribute__((ext_vector_type(8)));

// Build a WMMA f16 fragment from 16 rows (stride in halves) of LDS.
// Works for A (row = M, contiguous dim = K) and for column-major B
// (row = N column, contiguous dim = K).
// CDNA5 16-bit A layout: lane r=lane%16 holds row r; hi=lane/16 selects the
// +8 K offset half; VGPR v<4 -> K = hi*8 + 2v{+1}; v>=4 -> K = 16 + hi*8 + 2(v-4){+1}.
// Each 8-half group is contiguous -> lowers to two ds_load_b128.
__device__ __forceinline__ v16h frag_from_rows(const _Float16* p, int stride) {
  const int lane = threadIdx.x & 31;
  const int r = lane & 15, hi = lane >> 4;
  const _Float16* q = p + r * stride + hi * 8;
  v16h f;
#pragma unroll
  for (int v = 0; v < 4; ++v) {
    f[2 * v]         = q[2 * v];
    f[2 * v + 1]     = q[2 * v + 1];
    f[8 + 2 * v]     = q[16 + 2 * v];
    f[8 + 2 * v + 1] = q[16 + 2 * v + 1];
  }
  return f;
}

__device__ __forceinline__ v8f wmma_f16(v16h a, v16h b, v8f c) {
  return __builtin_amdgcn_wmma_f32_16x16x32_f16(false, a, false, b, (short)0, c,
                                                false, false);
}

// Async global -> LDS copy of 16 bytes (CDNA5, ASYNCcnt-tracked).
__device__ __forceinline__ void async_copy_b128(void* lds_ptr, const void* gptr) {
  uint32_t lds_off = (uint32_t)(uintptr_t)lds_ptr;  // low 32 bits = LDS offset
  asm volatile("global_load_async_to_lds_b128 %0, %1, off"
               :: "v"(lds_off), "v"(gptr)
               : "memory");
}
__device__ __forceinline__ void wait_async0() {
  asm volatile("s_wait_asynccnt 0x0" ::: "memory");
}

__device__ __forceinline__ float red16_max(float v) {
  v = fmaxf(v, __shfl_xor(v, 1, 32));
  v = fmaxf(v, __shfl_xor(v, 2, 32));
  v = fmaxf(v, __shfl_xor(v, 4, 32));
  v = fmaxf(v, __shfl_xor(v, 8, 32));
  return v;
}
__device__ __forceinline__ float red16_sum(float v) {
  v += __shfl_xor(v, 1, 32);
  v += __shfl_xor(v, 2, 32);
  v += __shfl_xor(v, 4, 32);
  v += __shfl_xor(v, 8, 32);
  return v;
}

// ---------------- Kernel 1: fused QKV projection -------------------------
// Y[m,e] = sum_k h[m,k] * W[e,k] + bias[e]; written f16 to [B,H,S,DH].
// Q gets the softmax scale (1/sqrt(DH)) folded in.
__global__ __launch_bounds__(256) void qkv_proj_kernel(
    const float* __restrict__ hsrc,
    const float* __restrict__ Wq, const float* __restrict__ bq,
    const float* __restrict__ Wk, const float* __restrict__ bk,
    const float* __restrict__ Wv, const float* __restrict__ bv,
    _Float16* __restrict__ Qo, _Float16* __restrict__ Ko,
    _Float16* __restrict__ Vo) {
  const int which = blockIdx.z;
  const float* W    = which == 0 ? Wq : (which == 1 ? Wk : Wv);
  const float* bias = which == 0 ? bq : (which == 1 ? bk : bv);
  _Float16* out     = which == 0 ? Qo : (which == 1 ? Ko : Vo);
  const float oscale = which == 0 ? 0.125f : 1.0f;  // 1/sqrt(64)

  const int m0 = blockIdx.y * 128;
  const int n0 = blockIdx.x * 64;
  const int tid = threadIdx.x;
  const int wave = tid >> 5;
  const int wm = (wave >> 1) * 32;  // 0..96
  const int wn = (wave & 1) * 32;   // 0,32

  __shared__ _Float16 As[128 * 40];
  __shared__ _Float16 Bs[64 * 40];

  v8f acc[2][2];
#pragma unroll
  for (int i = 0; i < 2; ++i)
#pragma unroll
    for (int j = 0; j < 2; ++j) acc[i][j] = (v8f){0, 0, 0, 0, 0, 0, 0, 0};

  for (int k0 = 0; k0 < D_; k0 += 32) {
    __syncthreads();
#pragma unroll
    for (int i = 0; i < 16; ++i) {  // 128x32 f32 -> f16
      int idx = tid + i * 256;
      int r = idx >> 5, c = idx & 31;
      As[r * 40 + c] = (_Float16)hsrc[(size_t)(m0 + r) * D_ + k0 + c];
    }
#pragma unroll
    for (int i = 0; i < 8; ++i) {   // 64 rows of W (column-major B)
      int idx = tid + i * 256;
      int r = idx >> 5, c = idx & 31;
      Bs[r * 40 + c] = (_Float16)W[(size_t)(n0 + r) * D_ + k0 + c];
    }
    __syncthreads();
    v16h a0 = frag_from_rows(&As[(wm + 0) * 40], 40);
    v16h a1 = frag_from_rows(&As[(wm + 16) * 40], 40);
    v16h b0 = frag_from_rows(&Bs[(wn + 0) * 40], 40);
    v16h b1 = frag_from_rows(&Bs[(wn + 16) * 40], 40);
    acc[0][0] = wmma_f16(a0, b0, acc[0][0]);
    acc[0][1] = wmma_f16(a0, b1, acc[0][1]);
    acc[1][0] = wmma_f16(a1, b0, acc[1][0]);
    acc[1][1] = wmma_f16(a1, b1, acc[1][1]);
  }

  const int lane = tid & 31;
  const int nloc = lane & 15, hi = lane >> 4;
#pragma unroll
  for (int sm = 0; sm < 2; ++sm)
#pragma unroll
    for (int sn = 0; sn < 2; ++sn) {
      int e = n0 + wn + sn * 16 + nloc;
      int hh = e >> 6, d = e & 63;
      float bsv = bias[e];
#pragma unroll
      for (int j = 0; j < 8; ++j) {
        int m = m0 + wm + sm * 16 + j + hi * 8;
        int bb = m >> 12;          // m / S_
        int s = m & (S_ - 1);
        size_t o = (((size_t)bb * H_ + hh) * S_ + s) * DH_ + d;
        out[o] = (_Float16)((acc[sm][sn][j] + bsv) * oscale);
      }
    }
}

// ---------------- Kernel 2: flash attention ------------------------------
#define QK_STRIDE 88   // halves; 176B rows -> B128-aligned, conflict-free reads
__global__ __launch_bounds__(128) void attn_kernel(
    const _Float16* __restrict__ Q, const _Float16* __restrict__ K,
    const _Float16* __restrict__ V, const int* __restrict__ mask,
    _Float16* __restrict__ att) {
  const int qt = blockIdx.x, hh = blockIdx.y, bb = blockIdx.z;
  const int tid = threadIdx.x, wave = tid >> 5, lane = tid & 31;
  const int nloc = lane & 15, hi = lane >> 4;

  __shared__ _Float16 Qs[64 * QK_STRIDE];
  __shared__ _Float16 Ks[64 * QK_STRIDE];
  __shared__ _Float16 Vt[64 * 72];        // transposed: [d][k]
  __shared__ _Float16 Ps[4][16 * 72];     // per-wave probabilities
  __shared__ float maskAdd[64];

  const size_t bh = ((size_t)bb * H_ + hh) * S_;

  {  // stage Q tile asynchronously (64 rows contiguous in memory)
    const _Float16* Qg = Q + (bh + (size_t)qt * 64) * DH_;
#pragma unroll
    for (int i = 0; i < 4; ++i) {
      int idx = tid + i * 128;            // B128 chunk index (8 halves)
      int r = idx >> 3, c8 = idx & 7;
      async_copy_b128(&Qs[r * QK_STRIDE + c8 * 8], Qg + idx * 8);
    }
  }
  wait_async0();
  __syncthreads();
  v16h aq0 = frag_from_rows(&Qs[(wave * 16) * QK_STRIDE + 0], QK_STRIDE);
  v16h aq1 = frag_from_rows(&Qs[(wave * 16) * QK_STRIDE + 32], QK_STRIDE);

  v8f o[4];
#pragma unroll
  for (int t = 0; t < 4; ++t) o[t] = (v8f){0, 0, 0, 0, 0, 0, 0, 0};
  float rmax[8], rsum[8];
#pragma unroll
  for (int j = 0; j < 8; ++j) { rmax[j] = -__builtin_inff(); rsum[j] = 0.f; }

  for (int k0 = 0; k0 < S_; k0 += 64) {
    __syncthreads();
    // K tile: async global -> LDS (no VGPR round trip)
    const _Float16* Kg = K + (bh + k0) * DH_;
#pragma unroll
    for (int i = 0; i < 4; ++i) {
      int idx = tid + i * 128;
      int r = idx >> 3, c8 = idx & 7;
      async_copy_b128(&Ks[r * QK_STRIDE + c8 * 8], Kg + idx * 8);
    }
    // V tile: sync load with transpose into [d][k]
    const uint2* vsrc = (const uint2*)(V + (bh + k0) * DH_);
#pragma unroll
    for (int i = 0; i < 8; ++i) {
      int idx = tid + i * 128;
      int r = idx >> 4, c4 = idx & 15;
      uint2 vv = vsrc[idx];
      const _Float16* vh = (const _Float16*)&vv;
#pragma unroll
      for (int u = 0; u < 4; ++u) Vt[(c4 * 4 + u) * 72 + r] = vh[u];
    }
    if (tid < 64) maskAdd[tid] = mask[bb * S_ + k0 + tid] ? -1e9f : 0.0f;
    wait_async0();
    __syncthreads();

    // prefetch next iteration's K/V lines while this tile computes
    if (k0 + 64 < S_) {
      __builtin_prefetch(K + (bh + k0 + 64) * DH_ + tid * 32, 0, 1);
      __builtin_prefetch(V + (bh + k0 + 64) * DH_ + tid * 32, 0, 1);
    }

    // scores: 16 queries x 64 keys, Q pre-scaled
    v8f c[4];
#pragma unroll
    for (int t = 0; t < 4; ++t) {
      v16h bk0 = frag_from_rows(&Ks[(t * 16) * QK_STRIDE + 0], QK_STRIDE);
      v16h bk1 = frag_from_rows(&Ks[(t * 16) * QK_STRIDE + 32], QK_STRIDE);
      v8f cc = (v8f){0, 0, 0, 0, 0, 0, 0, 0};
      cc = wmma_f16(aq0, bk0, cc);
      cc = wmma_f16(aq1, bk1, cc);
      float ma = maskAdd[t * 16 + nloc];
#pragma unroll
      for (int j = 0; j < 8; ++j) cc[j] += ma;
      c[t] = cc;
    }

    // online softmax (row m = j + 8*hi lives in one 16-lane half)
#pragma unroll
    for (int j = 0; j < 8; ++j) {
      float v = fmaxf(fmaxf(c[0][j], c[1][j]), fmaxf(c[2][j], c[3][j]));
      float tmax = red16_max(v);
      float nm = fmaxf(rmax[j], tmax);
      float corr = __expf(rmax[j] - nm);
      rmax[j] = nm;
#pragma unroll
      for (int t = 0; t < 4; ++t) c[t][j] = __expf(c[t][j] - nm);
      float s = red16_sum(c[0][j] + c[1][j] + c[2][j] + c[3][j]);
      rsum[j] = rsum[j] * corr + s;
#pragma unroll
      for (int t = 0; t < 4; ++t) o[t][j] *= corr;
    }

    // P -> LDS (A-fragment layout: rows = queries, contiguous = keys)
#pragma unroll
    for (int t = 0; t < 4; ++t)
#pragma unroll
      for (int j = 0; j < 8; ++j)
        Ps[wave][(j + hi * 8) * 72 + t * 16 + nloc] = (_Float16)c[t][j];
    __syncthreads();

    v16h ap0 = frag_from_rows(&Ps[wave][0], 72);    // keys 0..31
    v16h ap1 = frag_from_rows(&Ps[wave][32], 72);   // keys 32..63
#pragma unroll
    for (int t2 = 0; t2 < 4; ++t2) {
      v16h bv0 = frag_from_rows(&Vt[(t2 * 16) * 72 + 0], 72);
      v16h bv1 = frag_from_rows(&Vt[(t2 * 16) * 72 + 32], 72);
      o[t2] = wmma_f16(ap0, bv0, o[t2]);
      o[t2] = wmma_f16(ap1, bv1, o[t2]);
    }
  }

  // normalize + write att[b, s, h*64+d] as f16
  float inv[8];
#pragma unroll
  for (int j = 0; j < 8; ++j) inv[j] = 1.0f / rsum[j];
#pragma unroll
  for (int t2 = 0; t2 < 4; ++t2)
#pragma unroll
    for (int j = 0; j < 8; ++j) {
      size_t row = (size_t)bb * S_ + qt * 64 + wave * 16 + j + hi * 8;
      att[row * (H_ * DH_) + hh * DH_ + t2 * 16 + nloc] =
          (_Float16)(o[t2][j] * inv[j]);
    }
}

// ---------------- Kernel 3: output projection ----------------------------
// out[m,d] = sum_e att[m,e] * Wo[d,e] + bo[d]  (fp32 out)
__global__ __launch_bounds__(256) void out_proj_kernel(
    const _Float16* __restrict__ att, const float* __restrict__ Wo,
    const float* __restrict__ bo, float* __restrict__ out) {
  const int m0 = blockIdx.y * 128;
  const int n0 = blockIdx.x * 64;
  const int tid = threadIdx.x;
  const int wave = tid >> 5;
  const int wm = (wave >> 1) * 32;
  const int wn = (wave & 1) * 32;

  __shared__ _Float16 As[128 * 40];
  __shared__ _Float16 Bs[64 * 40];

  v8f acc[2][2];
#pragma unroll
  for (int i = 0; i < 2; ++i)
#pragma unroll
    for (int j = 0; j < 2; ++j) acc[i][j] = (v8f){0, 0, 0, 0, 0, 0, 0, 0};

  for (int k0 = 0; k0 < D_; k0 += 32) {
    __syncthreads();
#pragma unroll
    for (int i = 0; i < 8; ++i) {  // 128x32 halves as dwords
      int idx = tid + i * 256;
      int r = idx >> 4, cu = idx & 15;
      *(uint32_t*)&As[r * 40 + cu * 2] =
          ((const uint32_t*)att)[(((size_t)(m0 + r) * D_) + k0) / 2 + cu];
    }
#pragma unroll
    for (int i = 0; i < 8; ++i) {  // 64 rows of Wo, f32 -> f16
      int idx = tid + i * 256;
      int r = idx >> 5, c = idx & 31;
      Bs[r * 40 + c] = (_Float16)Wo[(size_t)(n0 + r) * D_ + k0 + c];
    }
    __syncthreads();
    v16h a0 = frag_from_rows(&As[(wm + 0) * 40], 40);
    v16h a1 = frag_from_rows(&As[(wm + 16) * 40], 40);
    v16h b0 = frag_from_rows(&Bs[(wn + 0) * 40], 40);
    v16h b1 = frag_from_rows(&Bs[(wn + 16) * 40], 40);
    acc[0][0] = wmma_f16(a0, b0, acc[0][0]);
    acc[0][1] = wmma_f16(a0, b1, acc[0][1]);
    acc[1][0] = wmma_f16(a1, b0, acc[1][0]);
    acc[1][1] = wmma_f16(a1, b1, acc[1][1]);
  }

  const int lane = tid & 31;
  const int nloc = lane & 15, hi = lane >> 4;
#pragma unroll
  for (int sm = 0; sm < 2; ++sm)
#pragma unroll
    for (int sn = 0; sn < 2; ++sn) {
      int e = n0 + wn + sn * 16 + nloc;
      float bsv = bo[e];
#pragma unroll
      for (int j = 0; j < 8; ++j) {
        int m = m0 + wm + sm * 16 + j + hi * 8;
        out[(size_t)m * D_ + e] = acc[sm][sn][j] + bsv;
      }
    }
}

extern "C" void kernel_launch(void* const* d_in, const int* in_sizes, int n_in,
                              void* d_out, int out_size, void* d_ws,
                              size_t ws_size, hipStream_t stream) {
  (void)in_sizes; (void)n_in; (void)out_size; (void)ws_size;
  const float* h  = (const float*)d_in[0];
  const float* Wq = (const float*)d_in[1];
  const float* bq = (const float*)d_in[2];
  const float* Wk = (const float*)d_in[3];
  const float* bk = (const float*)d_in[4];
  const float* Wv = (const float*)d_in[5];
  const float* bv = (const float*)d_in[6];
  const float* Wo = (const float*)d_in[7];
  const float* bo = (const float*)d_in[8];
  const int* mask = (const int*)d_in[9];
  float* out = (float*)d_out;

  _Float16* ws = (_Float16*)d_ws;
  const size_t per = (size_t)B_ * H_ * S_ * DH_;  // 6,291,456 halves
  _Float16* Qb = ws;
  _Float16* Kb = ws + per;
  _Float16* Vb = ws + 2 * per;
  _Float16* Ab = ws + 3 * per;

  qkv_proj_kernel<<<dim3(D_ / 64, (B_ * S_) / 128, 3), 256, 0, stream>>>(
      h, Wq, bq, Wk, bk, Wv, bv, Qb, Kb, Vb);
  attn_kernel<<<dim3(S_ / 64, H_, B_), 128, 0, stream>>>(Qb, Kb, Vb, mask, Ab);
  out_proj_kernel<<<dim3(D_ / 64, (B_ * S_) / 128), 256, 0, stream>>>(
      Ab, Wo, bo, out);
}
